// PointPillarScatter_Mix_full_54211077210392
// MI455X (gfx1250) — matrix operand order repaired and checked by reading, not compiled, not run on hardware
//
#include <hip/hip_runtime.h>

// ---------------- problem constants ----------------
#define NXc   432
#define NYc   496
#define NVc   4096
#define NPc   8192
#define Bc    4
#define NYNXc (NYc * NXc)      // 214272
#define EPSc  1e-3f

typedef __attribute__((ext_vector_type(16))) __bf16 v16bf;
typedef __attribute__((ext_vector_type(8)))  float  v8f;
typedef __attribute__((ext_vector_type(2)))  float  v2f;

#if __has_builtin(__builtin_amdgcn_wmma_f32_16x16x4_f32)
#define HAVE_WMMA_F32X4 1
#else
#define HAVE_WMMA_F32X4 0
#endif

// float -> bf16 (round-nearest-even), branchless
__device__ __forceinline__ unsigned f2bf(float f) {
  unsigned u = __float_as_uint(f);
  unsigned r = u + 0x7FFFu + ((u >> 16) & 1u);
  return r >> 16;
}

// pack 16 consecutive floats into a v16bf fragment register group
__device__ __forceinline__ v16bf pack16(const float* __restrict__ s) {
  union { v16bf v; unsigned u[8]; } r;
#pragma unroll
  for (int i = 0; i < 8; ++i)
    r.u[i] = f2bf(s[2 * i]) | (f2bf(s[2 * i + 1]) << 16);
  return r.v;
}

// branchless top-3 insert (v_cmp + v_cndmask only, no EXEC divergence)
__device__ __forceinline__ void top3_insert_bl(float val, int idx,
                                               float& v0, int& i0,
                                               float& v1, int& i1,
                                               float& v2, int& i2) {
  const bool g0 = val > v0;
  const bool g1 = val > v1;
  const bool g2 = val > v2;
  float nv2 = g1 ? v1 : (g2 ? val : v2);
  int   ni2 = g1 ? i1 : (g2 ? idx : i2);
  float nv1 = g0 ? v0 : (g1 ? val : v1);
  int   ni1 = g0 ? i0 : (g1 ? idx : i1);
  v0 = g0 ? val : v0;  i0 = g0 ? idx : i0;
  v1 = nv1; i1 = ni1;
  v2 = nv2; i2 = ni2;
}

__device__ __forceinline__ bool better(float a, int ai, float b, int bi) {
  return (a > b) || (a == b && ai < bi);
}

// branchless tie-aware insert (used only at the final lane-half merge)
__device__ __forceinline__ void top3_insert_tie(float val, int idx,
                                                float& v0, int& i0,
                                                float& v1, int& i1,
                                                float& v2, int& i2) {
  const bool g0 = better(val, idx, v0, i0);
  const bool g1 = better(val, idx, v1, i1);
  const bool g2 = better(val, idx, v2, i2);
  float nv2 = g1 ? v1 : (g2 ? val : v2);
  int   ni2 = g1 ? i1 : (g2 ? idx : i2);
  float nv1 = g0 ? v0 : (g1 ? val : v1);
  int   ni1 = g0 ? i0 : (g1 ? idx : i1);
  v0 = g0 ? val : v0;  i0 = g0 ? idx : i0;
  v1 = nv1; i1 = ni1;
  v2 = nv2; i2 = ni2;
}

// ---------------- zero-fill of the BEV output ----------------
__global__ void ppmix_zero_kernel(float* __restrict__ out, long long n) {
  long long i = (long long)blockIdx.x * blockDim.x + threadIdx.x;
  long long stride = (long long)gridDim.x * blockDim.x;
  long long n4 = n >> 2;
  float4 z; z.x = z.y = z.z = z.w = 0.f;
  for (long long j = i; j < n4; j += stride) ((float4*)out)[j] = z;
  for (long long j = (n4 << 2) + i; j < n; j += stride) out[j] = 0.f;
}

// ---------------- fused scoring + mix + adapter + scatter ----------------
// block = 256 threads = 8 waves; each wave owns 16 pillars; block owns 128.
__global__ __launch_bounds__(256)
void ppmix_fused_kernel(const float* __restrict__ pf,   // [B,NV,64]
                        const float* __restrict__ ptf,  // [B,NP,64]
                        const float* __restrict__ vc,   // [B,NV,4]
                        const float* __restrict__ pc,   // [B,NP,4]
                        const float* __restrict__ Wa,   // [64,192]
                        const float* __restrict__ Ww,   // [2,64]
                        const float* __restrict__ g1, const float* __restrict__ bt1,
                        const float* __restrict__ mn1, const float* __restrict__ vr1,
                        const float* __restrict__ g2, const float* __restrict__ bt2,
                        const float* __restrict__ mn2, const float* __restrict__ vr2,
                        float* __restrict__ out) {
  const int b    = blockIdx.y;
  const int tid  = threadIdx.x;
  const int wave = tid >> 5;
  const int ln   = tid & 31;
  const int col  = ln & 15;   // column within wave tile (pillar / out-channel)
  const int hh   = ln >> 4;   // lane-half
  const int vBase = blockIdx.x * 128 + wave * 16;
  const int v     = vBase + col;                // this lane's pillar

  __shared__ unsigned short ldsPts[2][32 * 64];   // double-buffered 32x64 bf16 point tile
  __shared__ unsigned short ldsPP[8][16 * 192];   // per-wave mixed features (bf16)
  __shared__ int   ldsLin[8][16];
  __shared__ float ldsW[8][16][2];

  // ---- B fragments: pillar features (K x N, K=64 split into 2 chunks of 32) ----
  const float* pfRow = pf + ((size_t)b * NVc + v) * 64;
  v16bf Bf0 = pack16(pfRow + 0 * 32 + 16 * hh);
  v16bf Bf1 = pack16(pfRow + 1 * 32 + 16 * hh);

  // coord B fragment for 16x16x4 f32 WMMA (lane supplies K = 2h, 2h+1)
  const float* vcRow = vc + ((size_t)b * NVc + v) * 4;
#if HAVE_WMMA_F32X4
  v2f Bc2 = { vcRow[2 * hh], vcRow[2 * hh + 1] };
#else
  float4 vcAll = *(const float4*)vcRow;
#endif

  // running top-3 (feature scores s1, coord scores s2)
  float s1v0 = -INFINITY, s1v1 = -INFINITY, s1v2 = -INFINITY;
  float s2v0 = -INFINITY, s2v1 = -INFINITY, s2v2 = -INFINITY;
  int   s1i0 = 0x7fffffff, s1i1 = 0x7fffffff, s1i2 = 0x7fffffff;
  int   s2i0 = 0x7fffffff, s2i1 = 0x7fffffff, s2i2 = 0x7fffffff;

  const float* ptfB = ptf + (size_t)b * NPc * 64;

  // stage a 32x64 fp32 point tile -> bf16 into LDS buffer `bf`
  auto stage = [&](int tt, int bf) {
    const float4* src = (const float4*)(ptfB + (size_t)tt * 32 * 64);
    uint2* dst = (uint2*)ldsPts[bf];
#pragma unroll
    for (int k = 0; k < 2; ++k) {
      float4 d = src[tid + k * 256];
      uint2 t2;
      t2.x = f2bf(d.x) | (f2bf(d.y) << 16);
      t2.y = f2bf(d.z) | (f2bf(d.w) << 16);
      dst[tid + k * 256] = t2;
    }
  };

  const int NT = NPc / 32;  // 256 tiles of 32 points
  stage(0, 0);
  for (int t = 0; t < NT; ++t) {
    __syncthreads();
    const int cur = t & 1;
    if (t + 1 < NT) stage(t + 1, cur ^ 1);
    if (t + 2 < NT)
      __builtin_prefetch(ptfB + (size_t)(t + 2) * 32 * 64 + tid * 8, 0, 3);

#pragma unroll
    for (int s = 0; s < 2; ++s) {
      // ---- A fragments from LDS (16-bit A 16x32 lane layout) ----
      union AU { v16bf v; uint4 q[2]; } a0, a1;
      const unsigned short* rowp = ldsPts[cur] + (s * 16 + col) * 64;
      a0.q[0] = *(const uint4*)(rowp + 0 * 32 + 16 * hh);
      a0.q[1] = *(const uint4*)(rowp + 0 * 32 + 16 * hh + 8);
      a1.q[0] = *(const uint4*)(rowp + 1 * 32 + 16 * hh);
      a1.q[1] = *(const uint4*)(rowp + 1 * 32 + 16 * hh + 8);

      // ---- feature scores: D = A(points 16xK) x B(pillars Kx16), K=64 ----
      v8f acc = {0.f, 0.f, 0.f, 0.f, 0.f, 0.f, 0.f, 0.f};
      acc = __builtin_amdgcn_wmma_f32_16x16x32_bf16(false, a0.v, false, Bf0,
                                                    (short)0, acc, false, false);
      acc = __builtin_amdgcn_wmma_f32_16x16x32_bf16(false, a1.v, false, Bf1,
                                                    (short)0, acc, false, false);

      // ---- coord scores (K=4) ----
      v8f acc2 = {0.f, 0.f, 0.f, 0.f, 0.f, 0.f, 0.f, 0.f};
#if HAVE_WMMA_F32X4
      {
        const float* pcRow = pc + ((size_t)b * NPc + t * 32 + s * 16 + col) * 4 + 2 * hh;
        v2f Ac = { pcRow[0], pcRow[1] };
        acc2 = __builtin_amdgcn_wmma_f32_16x16x4_f32(false, Ac, false, Bc2,
                                                     (short)0, acc2, false, false);
      }
#else
#pragma unroll
      for (int r = 0; r < 8; ++r) {
        int p = t * 32 + s * 16 + r + 8 * hh;
        float4 pr = *(const float4*)(pc + ((size_t)b * NPc + p) * 4);
        acc2[r] = pr.x * vcAll.x + pr.y * vcAll.y + pr.z * vcAll.z + pr.w * vcAll.w;
      }
#endif

      // ---- running top-3 update (C/D layout: lane holds col, rows r+8*hh) ----
      const int pbase = t * 32 + s * 16 + 8 * hh;
      float bm1 = acc[0], bm2 = acc2[0];
#pragma unroll
      for (int r = 1; r < 8; ++r) { bm1 = fmaxf(bm1, acc[r]); bm2 = fmaxf(bm2, acc2[r]); }
      if (__any(bm1 > s1v2)) {   // wave-uniform skip: most batches cannot improve top-3
#pragma unroll
        for (int r = 0; r < 8; ++r)
          top3_insert_bl(acc[r], pbase + r, s1v0, s1i0, s1v1, s1i1, s1v2, s1i2);
      }
      if (__any(bm2 > s2v2)) {
#pragma unroll
        for (int r = 0; r < 8; ++r)
          top3_insert_bl(acc2[r], pbase + r, s2v0, s2i0, s2v1, s2i1, s2v2, s2i2);
      }
    }
  }

  // ---- merge the two lane halves (each saw half of the point rows) ----
  {
    float o0 = __shfl_xor(s1v0, 16), o1 = __shfl_xor(s1v1, 16), o2 = __shfl_xor(s1v2, 16);
    int   j0 = __shfl_xor(s1i0, 16), j1 = __shfl_xor(s1i1, 16), j2 = __shfl_xor(s1i2, 16);
    top3_insert_tie(o0, j0, s1v0, s1i0, s1v1, s1i1, s1v2, s1i2);
    top3_insert_tie(o1, j1, s1v0, s1i0, s1v1, s1i1, s1v2, s1i2);
    top3_insert_tie(o2, j2, s1v0, s1i0, s1v1, s1i1, s1v2, s1i2);
    o0 = __shfl_xor(s2v0, 16); o1 = __shfl_xor(s2v1, 16); o2 = __shfl_xor(s2v2, 16);
    j0 = __shfl_xor(s2i0, 16); j1 = __shfl_xor(s2i1, 16); j2 = __shfl_xor(s2i2, 16);
    top3_insert_tie(o0, j0, s2v0, s2i0, s2v1, s2i1, s2v2, s2i2);
    top3_insert_tie(o1, j1, s2v0, s2i0, s2v1, s2i1, s2v2, s2i2);
    top3_insert_tie(o2, j2, s2v0, s2i0, s2v1, s2i1, s2v2, s2i2);
  }

  // ================= tail =================
  // T1: linear BEV index, gating weights, pind planes, pf passthrough
  const float zf = vcRow[1], yf = vcRow[2], xf = vcRow[3];
  const int lin = (int)(zf + yf * (float)NXc + xf);
  if (hh == 0) ldsLin[wave][col] = lin;

  {
    // gating logit: lane (col, hh) computes channel hh for pillar col
    const float* wwRow = Ww + hh * 64;
    float dot = 0.f;
#pragma unroll
    for (int k = 0; k < 64; k += 4) {
      float4 a = *(const float4*)(pfRow + k);
      float4 w = *(const float4*)(wwRow + k);
      dot += a.x * w.x + a.y * w.y + a.z * w.z + a.w * w.w;
    }
    float z = (dot - mn2[hh]) * (g2[hh] * rsqrtf(vr2[hh] + EPSc)) + bt2[hh];
    float zo = __shfl_xor(z, 16);
    float mx = fmaxf(z, zo);
    float e = __expf(z - mx), eo = __expf(zo - mx);
    ldsW[wave][col][hh] = e / (e + eo);
  }

  {
    // pind planes (y, x, z) and pf passthrough channels
    const size_t idxBase = (size_t)Bc * 128 * NYNXc;
    if (hh == 0) {
      float* oi = out + idxBase + ((size_t)b * 3) * NYNXc + lin;
      oi[0 * (size_t)NYNXc] = yf;
      oi[1 * (size_t)NYNXc] = xf;
      oi[2 * (size_t)NYNXc] = zf;
    }
#pragma unroll
    for (int c0 = 0; c0 < 32; c0 += 4) {
      float4 f = *(const float4*)(pfRow + 32 * hh + c0);
      size_t cb = (size_t)b * 128 + 32 * hh + c0;
      out[(cb + 0) * NYNXc + lin] = f.x;
      out[(cb + 1) * NYNXc + lin] = f.y;
      out[(cb + 2) * NYNXc + lin] = f.z;
      out[(cb + 3) * NYNXc + lin] = f.w;
    }
  }
  __syncthreads();

  // T2: build pp = w0*ptf[idx1] + w1*ptf[idx2] -> bf16 in LDS
  {
    const float w0 = ldsW[wave][col][0];
    const float w1 = ldsW[wave][col][1];
    const int i1a[3] = { s1i0, s1i1, s1i2 };
    const int i2a[3] = { s2i0, s2i1, s2i2 };
#pragma unroll
    for (int kk = 0; kk < 3; ++kk) {
      const float* r1 = ptfB + (size_t)i1a[kk] * 64 + 32 * hh;
      const float* r2 = ptfB + (size_t)i2a[kk] * 64 + 32 * hh;
#pragma unroll
      for (int c0 = 0; c0 < 32; c0 += 4) {
        float4 f1 = *(const float4*)(r1 + c0);
        float4 f2 = *(const float4*)(r2 + c0);
        float x0 = w0 * f1.x + w1 * f2.x, x1 = w0 * f1.y + w1 * f2.y;
        float x2 = w0 * f1.z + w1 * f2.z, x3 = w0 * f1.w + w1 * f2.w;
        uint2 t2;
        t2.x = f2bf(x0) | (f2bf(x1) << 16);
        t2.y = f2bf(x2) | (f2bf(x3) << 16);
        *(uint2*)&ldsPP[wave][col * 192 + kk * 64 + 32 * hh + c0] = t2;
      }
    }
  }
  __syncthreads();

  // T3: adapter GEMM via WMMA: D(16 pillars x 64ch) = pp(16x192) x Wa^T(192x64)
  {
    union AU { v16bf v; uint4 q[2]; } af[6];
#pragma unroll
    for (int qk = 0; qk < 6; ++qk) {
      const unsigned short* base = &ldsPP[wave][col * 192 + qk * 32 + 16 * hh];
      af[qk].q[0] = *(const uint4*)(base);
      af[qk].q[1] = *(const uint4*)(base + 8);
    }
#pragma unroll
    for (int nt = 0; nt < 4; ++nt) {
      v8f acc = {0.f, 0.f, 0.f, 0.f, 0.f, 0.f, 0.f, 0.f};
      const int c = nt * 16 + col;
      const float* wrow = Wa + (size_t)c * 192;
#pragma unroll
      for (int qk = 0; qk < 6; ++qk) {
        v16bf bfr = pack16(wrow + qk * 32 + 16 * hh);
        acc = __builtin_amdgcn_wmma_f32_16x16x32_bf16(false, af[qk].v, false, bfr,
                                                      (short)0, acc, false, false);
      }
      const float sc = g1[c] * rsqrtf(vr1[c] + EPSc);
      const float mu = mn1[c], be = bt1[c];
#pragma unroll
      for (int r = 0; r < 8; ++r) {
        const int row = r + 8 * hh;              // pillar within wave tile
        const int lin2 = ldsLin[wave][row];
        float val = (acc[r] - mu) * sc + be;
        val = fmaxf(val, 0.f);
        out[((size_t)b * 128 + 64 + c) * NYNXc + lin2] = val;
      }
    }
  }
}

// ---------------- launch ----------------
extern "C" void kernel_launch(void* const* d_in, const int* in_sizes, int n_in,
                              void* d_out, int out_size, void* d_ws, size_t ws_size,
                              hipStream_t stream) {
  (void)in_sizes; (void)n_in; (void)d_ws; (void)ws_size;
  const float* pf  = (const float*)d_in[0];
  const float* ptf = (const float*)d_in[1];
  const float* vcp = (const float*)d_in[2];
  const float* pcp = (const float*)d_in[3];
  const float* Wa  = (const float*)d_in[4];
  const float* Ww  = (const float*)d_in[5];
  const float* g1  = (const float*)d_in[6];
  const float* bt1 = (const float*)d_in[7];
  const float* mn1 = (const float*)d_in[8];
  const float* vr1 = (const float*)d_in[9];
  const float* g2  = (const float*)d_in[10];
  const float* bt2 = (const float*)d_in[11];
  const float* mn2 = (const float*)d_in[12];
  const float* vr2 = (const float*)d_in[13];
  float* out = (float*)d_out;

  ppmix_zero_kernel<<<4096, 256, 0, stream>>>(out, (long long)out_size);

  dim3 grid(NVc / 128, Bc);
  ppmix_fused_kernel<<<grid, 256, 0, stream>>>(pf, ptf, vcp, pcp, Wa, Ww,
                                               g1, bt1, mn1, vr1,
                                               g2, bt2, mn2, vr2, out);
}
